// LSTMAggregator_62775241999126
// MI455X (gfx1250) — matrix operand order repaired and checked by reading, not compiled, not run on hardware
//
#include <hip/hip_runtime.h>

typedef __attribute__((ext_vector_type(16))) __bf16 v16bf;
typedef __attribute__((ext_vector_type(8)))  __bf16 v8bf;
typedef __attribute__((ext_vector_type(8)))  float  v8f;

#define DI __device__ __forceinline__

DI float sigf(float x) { return 1.0f / (1.0f + __expf(-x)); }

// Pack 8 floats (two float4) into a v8bf and store (16B) to LDS.
DI void pack8(const float4& f0, const float4& f1, __bf16* dst) {
  v8bf v;
  v[0] = (__bf16)f0.x; v[1] = (__bf16)f0.y; v[2] = (__bf16)f0.z; v[3] = (__bf16)f0.w;
  v[4] = (__bf16)f1.x; v[5] = (__bf16)f1.y; v[6] = (__bf16)f1.z; v[7] = (__bf16)f1.w;
  *(v8bf*)dst = v;
}

// Build a 16x32 bf16 A-fragment from a row-major bf16 row in LDS.
// Lane (m = lane&15, kh = lane>>4): elems 0..15 = K base+kh*16..
DI v16bf load_afrag(const __bf16* row_base /* = &lds[m*stride + kt*32] */, int kh) {
  v8bf lo = *(const v8bf*)(row_base + kh * 16);
  v8bf hi = *(const v8bf*)(row_base + kh * 16 + 8);
  return __builtin_shufflevector(lo, hi, 0, 1, 2, 3, 4, 5, 6, 7, 8, 9, 10, 11, 12, 13, 14, 15);
}

// ---------------- weight convert: f32 -> bf16 ----------------
__global__ void cvt_kernel(const float* __restrict__ src, __bf16* __restrict__ dst, int n) {
  int i = blockIdx.x * blockDim.x + threadIdx.x;
  if (i < n) dst[i] = (__bf16)src[i];
}

// ---------------- fused one-direction LSTM over a 32-node tile ----------------
// grid: (313, 2); block: 256 threads (8 waves). Dynamic LDS:
//   s_wih [512][256] bf16 (256KB) | s_x [32][256] bf16 (16KB) | s_h [32][128] bf16 (8KB)
__global__ __launch_bounds__(256) void lstm_kernel(
    const float* __restrict__ nb,                                  // [10000,32,256]
    const __bf16* __restrict__ wih_f, const __bf16* __restrict__ whh_f, const float* __restrict__ b_f,
    const __bf16* __restrict__ wih_b, const __bf16* __restrict__ whh_b, const float* __restrict__ b_b,
    float* __restrict__ h_agg)                                     // [10000,256]
{
  extern __shared__ char smem[];
  __bf16* s_wih = (__bf16*)smem;                                   // 512*256
  __bf16* s_x   = (__bf16*)(smem + 512 * 256 * 2);                 // 32*256
  __bf16* s_h   = (__bf16*)(smem + 512 * 256 * 2 + 32 * 256 * 2);  // 32*128

  const int tile = blockIdx.x;     // 32-node tile
  const int dir  = blockIdx.y;
  const int tid  = threadIdx.x;
  const int wv   = tid >> 5;       // wave 0..7 -> owns hidden cols [wv*16, wv*16+16)
  const int ln   = tid & 31;
  const int lnL  = ln & 15;        // A: row M (within M-tile); B/C/D: col N
  const int lnH  = ln >> 4;        // A/B: K-half; C/D: M-half

  const __bf16* wih = dir ? wih_b : wih_f;
  const __bf16* whh = dir ? whh_b : whh_f;
  const float*  bia = dir ? b_b : b_f;

  // cooperative x loader roles: row m = tid>>3 (0..31), 32-col chunk ch = tid&7
  const int xm = tid >> 3, xc = tid & 7;
  const int xnode = tile * 32 + xm;
  const int xnclamp = xnode < 10000 ? xnode : 9999;  // last tile: clamp (rows unused)
  const float* xbase = nb + (size_t)xnclamp * 32 * 256 + xc * 32;

  // prefetch x[0] into registers (overlaps with w_ih staging below)
  float4 xf[8];
  {
    const float4* p0 = (const float4*)(xbase + (size_t)(dir ? 31 : 0) * 256);
#pragma unroll
    for (int q = 0; q < 8; ++q) xf[q] = p0[q];
  }

  // stage w_ih bf16 into LDS (16B chunks: 16384 total, 64 per thread)
  {
    const uint4* s = (const uint4*)wih;
    uint4* d = (uint4*)s_wih;
    for (int i = tid; i < (512 * 256 * 2) / 16; i += 256) d[i] = s[i];
  }
  for (int i = tid; i < 32 * 128; i += 256) s_h[i] = (__bf16)0.0f;

  // resident recurrent-weight B-fragments: gate g -> N-tile rows [g*128 + wv*16 ...)
  v16bf bhh[4][4];
#pragma unroll
  for (int g = 0; g < 4; ++g)
#pragma unroll
    for (int kt = 0; kt < 4; ++kt)
      bhh[g][kt] = *(const v16bf*)&whh[(size_t)(g * 128 + wv * 16 + lnL) * 128 + kt * 32 + lnH * 16];

  float bv[4];
#pragma unroll
  for (int g = 0; g < 4; ++g) bv[g] = bia[g * 128 + wv * 16 + lnL];

  float c[2][8], h[2][8];
#pragma unroll
  for (int mt = 0; mt < 2; ++mt)
#pragma unroll
    for (int r = 0; r < 8; ++r) { c[mt][r] = 0.0f; h[mt][r] = 0.0f; }

  __syncthreads();  // s_wih staged, s_h zeroed

  for (int t = 0; t < 32; ++t) {
    // ---- publish prefetched x[t]: regs -> bf16 LDS (disjoint from s_h writes) ----
    pack8(xf[0], xf[1], &s_x[xm * 256 + xc * 32]);
    pack8(xf[2], xf[3], &s_x[xm * 256 + xc * 32 + 8]);
    pack8(xf[4], xf[5], &s_x[xm * 256 + xc * 32 + 16]);
    pack8(xf[6], xf[7], &s_x[xm * 256 + xc * 32 + 24]);
    __syncthreads();  // [A] s_x[t] and s_h[t-1] visible to all waves

    // ---- issue prefetch of x[t+1]; drains behind the WMMA phase ----
    if (t < 31) {
      const int tn = dir ? (30 - t) : (t + 1);
      const float4* pn = (const float4*)(xbase + (size_t)tn * 256);
#pragma unroll
      for (int q = 0; q < 8; ++q) xf[q] = pn[q];
    }

    v8f acc[2][4];
#pragma unroll
    for (int mt = 0; mt < 2; ++mt)
#pragma unroll
      for (int g = 0; g < 4; ++g)
#pragma unroll
        for (int r = 0; r < 8; ++r) acc[mt][g][r] = bv[g];

    // ---- gates += x @ w_ih^T  (K = 256 -> 8 K-tiles; each B-frag feeds 2 M-tiles) ----
#pragma unroll
    for (int kt = 0; kt < 8; ++kt) {
      v16bf a0 = load_afrag(&s_x[lnL * 256 + kt * 32], lnH);
      v16bf a1 = load_afrag(&s_x[(16 + lnL) * 256 + kt * 32], lnH);
#pragma unroll
      for (int g = 0; g < 4; ++g) {
        v16bf b = *(const v16bf*)&s_wih[(size_t)(g * 128 + wv * 16 + lnL) * 256 + kt * 32 + lnH * 16];
        acc[0][g] = __builtin_amdgcn_wmma_f32_16x16x32_bf16(false, a0, false, b, (short)0, acc[0][g], false, false);
        acc[1][g] = __builtin_amdgcn_wmma_f32_16x16x32_bf16(false, a1, false, b, (short)0, acc[1][g], false, false);
      }
    }
    // ---- gates += h @ w_hh^T  (K = 128 -> 4 K-tiles, weights in VGPRs) ----
#pragma unroll
    for (int kt = 0; kt < 4; ++kt) {
      v16bf a0 = load_afrag(&s_h[lnL * 128 + kt * 32], lnH);
      v16bf a1 = load_afrag(&s_h[(16 + lnL) * 128 + kt * 32], lnH);
#pragma unroll
      for (int g = 0; g < 4; ++g) {
        acc[0][g] = __builtin_amdgcn_wmma_f32_16x16x32_bf16(false, a0, false, bhh[g][kt], (short)0, acc[0][g], false, false);
        acc[1][g] = __builtin_amdgcn_wmma_f32_16x16x32_bf16(false, a1, false, bhh[g][kt], (short)0, acc[1][g], false, false);
      }
    }

    // ---- LSTM cell update (wave-local: i/f/g/o tiles share columns) ----
#pragma unroll
    for (int mt = 0; mt < 2; ++mt)
#pragma unroll
      for (int r = 0; r < 8; ++r) {
        float iv = sigf(acc[mt][0][r]);
        float fv = sigf(acc[mt][1][r]);
        float gv = tanhf(acc[mt][2][r]);
        float ov = sigf(acc[mt][3][r]);
        float cv = fv * c[mt][r] + iv * gv;
        c[mt][r] = cv;
        h[mt][r] = ov * tanhf(cv);
      }

    __syncthreads();  // [B] all waves done reading s_x[t] / s_h[t-1]
#pragma unroll
    for (int mt = 0; mt < 2; ++mt)
#pragma unroll
      for (int r = 0; r < 8; ++r)
        s_h[(mt * 16 + lnH * 8 + r) * 128 + wv * 16 + lnL] = (__bf16)h[mt][r];
    // published by barrier [A] of next iteration
  }

  // final hidden state -> h_agg[node][dir*128 + col]  (f32)
#pragma unroll
  for (int mt = 0; mt < 2; ++mt)
#pragma unroll
    for (int r = 0; r < 8; ++r) {
      int node = tile * 32 + mt * 16 + lnH * 8 + r;
      if (node < 10000)
        h_agg[(size_t)node * 256 + dir * 128 + wv * 16 + lnL] = h[mt][r];
    }
}

// ---------------- final linear: relu([nf|h_agg] @ w_lin^T + b) ----------------
// grid: 625; block 256 (8 waves); each wave owns out-col tiles {wv, wv+8}.
__global__ __launch_bounds__(256) void linear_kernel(
    const float* __restrict__ nf, const float* __restrict__ hagg,
    const __bf16* __restrict__ wlin,   // [256][512] bf16 row-major
    const float* __restrict__ blin, float* __restrict__ out)
{
  __shared__ __bf16 s_a[16 * 512];
  const int tile = blockIdx.x, tid = threadIdx.x;
  const int wv = tid >> 5, ln = tid & 31, lnL = ln & 15, lnH = ln >> 4;

  // stage combined [16, 512] bf16: cols 0..255 = node_feature, 256..511 = h_agg
  {
    const int m = tid >> 4, ch = tid & 15;
    const float* src = (ch < 8) ? (nf + (size_t)(tile * 16 + m) * 256 + ch * 32)
                                : (hagg + (size_t)(tile * 16 + m) * 256 + (ch - 8) * 32);
#pragma unroll
    for (int q = 0; q < 2; ++q) {
      const float4* s4 = (const float4*)(src + q * 16);
      float4 f0 = s4[0], f1 = s4[1], f2 = s4[2], f3 = s4[3];
      pack8(f0, f1, &s_a[m * 512 + ch * 32 + q * 16]);
      pack8(f2, f3, &s_a[m * 512 + ch * 32 + q * 16 + 8]);
    }
  }
  __syncthreads();

  v8f acc[2];
#pragma unroll
  for (int u = 0; u < 2; ++u) {
    float b = blin[u * 128 + wv * 16 + lnL];
#pragma unroll
    for (int r = 0; r < 8; ++r) acc[u][r] = b;
  }

#pragma unroll
  for (int kt = 0; kt < 16; ++kt) {
    v16bf a = load_afrag(&s_a[lnL * 512 + kt * 32], lnH);
#pragma unroll
    for (int u = 0; u < 2; ++u) {
      v16bf b = *(const v16bf*)&wlin[(size_t)(u * 128 + wv * 16 + lnL) * 512 + kt * 32 + lnH * 16];
      acc[u] = __builtin_amdgcn_wmma_f32_16x16x32_bf16(false, a, false, b, (short)0, acc[u], false, false);
    }
  }

#pragma unroll
  for (int u = 0; u < 2; ++u)
#pragma unroll
    for (int r = 0; r < 8; ++r) {
      int node = tile * 16 + lnH * 8 + r;
      int col = u * 128 + wv * 16 + lnL;
      out[(size_t)node * 256 + col] = fmaxf(acc[u][r], 0.0f);
    }
}

extern "C" void kernel_launch(void* const* d_in, const int* in_sizes, int n_in,
                              void* d_out, int out_size, void* d_ws, size_t ws_size,
                              hipStream_t stream) {
  const float* nf   = (const float*)d_in[0];
  const float* nb   = (const float*)d_in[1];
  const float* wihf = (const float*)d_in[2];
  const float* whhf = (const float*)d_in[3];
  const float* bf   = (const float*)d_in[4];
  const float* wihb = (const float*)d_in[5];
  const float* whhb = (const float*)d_in[6];
  const float* bb   = (const float*)d_in[7];
  const float* wlin = (const float*)d_in[8];
  const float* blin = (const float*)d_in[9];
  float* out = (float*)d_out;

  char* ws = (char*)d_ws;
  __bf16* wihf16 = (__bf16*)(ws + 0);        // 512*256 bf16 = 256KB
  __bf16* wihb16 = (__bf16*)(ws + 262144);   // 256KB
  __bf16* whhf16 = (__bf16*)(ws + 524288);   // 512*128 bf16 = 128KB
  __bf16* whhb16 = (__bf16*)(ws + 655360);   // 128KB
  __bf16* wlin16 = (__bf16*)(ws + 786432);   // 256*512 bf16 = 256KB
  float*  hagg   = (float*)(ws + 1048576);   // 10000*256 f32 = 10MB

  cvt_kernel<<<512, 256, 0, stream>>>(wihf, wihf16, 512 * 256);
  cvt_kernel<<<512, 256, 0, stream>>>(wihb, wihb16, 512 * 256);
  cvt_kernel<<<256, 256, 0, stream>>>(whhf, whhf16, 512 * 128);
  cvt_kernel<<<256, 256, 0, stream>>>(whhb, whhb16, 512 * 128);
  cvt_kernel<<<512, 256, 0, stream>>>(wlin, wlin16, 256 * 512);

  const int smem = 512 * 256 * 2 + 32 * 256 * 2 + 32 * 128 * 2;  // 286720 B (< 320KB WGP LDS)
  (void)hipFuncSetAttribute((const void*)lstm_kernel,
                            hipFuncAttributeMaxDynamicSharedMemorySize, smem);
  lstm_kernel<<<dim3(313, 2), 256, smem, stream>>>(nb, wihf16, whhf16, bf,
                                                   wihb16, whhb16, bb, hagg);
  linear_kernel<<<625, 256, 0, stream>>>(nf, hagg, wlin16, blin, out);
}
